// ConvLayer_90099823935628
// MI455X (gfx1250) — matrix operand order repaired and checked by reading, not compiled, not run on hardware
//
#include <hip/hip_runtime.h>
#include <hip/hip_bf16.h>

#define N_NODES 50000
#define N_EDGES 150000
#define H       512
#define H4      2048
#define LN_EPS  1e-5f

typedef __attribute__((ext_vector_type(16))) __bf16 v16bf;
typedef __attribute__((ext_vector_type(8)))  __bf16 v8bf;
typedef __attribute__((ext_vector_type(4)))  __bf16 v4bf;
typedef __attribute__((ext_vector_type(8)))  float  v8f;

// ---------------------------------------------------------------------------
// Edge scatter: aggr[dst] += relu(x[src] + edge_attr), float4-vectorized.
// ---------------------------------------------------------------------------
__global__ __launch_bounds__(256) void k_scatter(const float* __restrict__ x,
                                                 const float* __restrict__ ea,
                                                 const int*   __restrict__ ei,
                                                 float*       __restrict__ aggr) {
    long tid = (long)blockIdx.x * blockDim.x + threadIdx.x;
    if (tid >= (long)N_EDGES * (H / 4)) return;
    int e = (int)(tid >> 7);           // H/4 == 128
    int c = ((int)tid & 127) << 2;
    int src = ei[e];
    int dst = ei[N_EDGES + e];
    float4 xv = *(const float4*)(x  + (size_t)src * H + c);
    float4 av = *(const float4*)(ea + (size_t)e   * H + c);
    float4 m;
    m.x = fmaxf(xv.x + av.x, 0.f);
    m.y = fmaxf(xv.y + av.y, 0.f);
    m.z = fmaxf(xv.z + av.z, 0.f);
    m.w = fmaxf(xv.w + av.w, 0.f);
    float* p = aggr + (size_t)dst * H + c;
    unsafeAtomicAdd(p + 0, m.x);
    unsafeAtomicAdd(p + 1, m.y);
    unsafeAtomicAdd(p + 2, m.z);
    unsafeAtomicAdd(p + 3, m.w);
}

// ---------------------------------------------------------------------------
// h_bf16 = bf16(x + aggr)   (GINE eps = 0)
// ---------------------------------------------------------------------------
__global__ __launch_bounds__(256) void k_makeh(const float* __restrict__ x,
                                               const float* __restrict__ aggr,
                                               __bf16*      __restrict__ h) {
    long tid = (long)blockIdx.x * blockDim.x + threadIdx.x;
    if (tid >= (long)N_NODES * (H / 4)) return;
    size_t i = (size_t)tid * 4;
    float4 xv = *(const float4*)(x + i);
    float4 av = *(const float4*)(aggr + i);
    v4bf o;
    o[0] = (__bf16)(xv.x + av.x);
    o[1] = (__bf16)(xv.y + av.y);
    o[2] = (__bf16)(xv.z + av.z);
    o[3] = (__bf16)(xv.w + av.w);
    *(v4bf*)(h + i) = o;
}

// ---------------------------------------------------------------------------
// Transpose + convert weights: WT[c][r] = bf16(W[r][c]),  W is rows x cols.
// ---------------------------------------------------------------------------
__global__ __launch_bounds__(256) void k_packT(const float* __restrict__ W,
                                               __bf16*      __restrict__ WT,
                                               int rows, int cols) {
    long tid = (long)blockIdx.x * blockDim.x + threadIdx.x;
    if (tid >= (long)rows * cols) return;
    int r = (int)(tid / cols);
    int c = (int)(tid % cols);
    WT[(size_t)c * rows + r] = (__bf16)W[tid];
}

// ---------------------------------------------------------------------------
// A-matrix fragment (16x32 bf16) from a row-major row pointer (already at k0).
// Layout: lane half h holds K[16h..16h+15] contiguously (h = lane>>4).
// ---------------------------------------------------------------------------
__device__ __forceinline__ v16bf load_a_frag(const __bf16* rowAtK0, int lh) {
    union { v16bf v; v8bf h[2]; } u;
    u.h[0] = *(const v8bf*)(rowAtK0 + 16 * lh);
    u.h[1] = *(const v8bf*)(rowAtK0 + 16 * lh + 8);
    return u.v;
}

// ---------------------------------------------------------------------------
// Fused MLP: per block = 16 node rows, 8 waves.
//   Phase 1: hid[m][n] = gelu((x+aggr) @ W1)  via D = W1T_tile x h^T_tile
//   Phase 2: out_pre   = x + hid @ W2         via D = W2T_tile x hid^T_tile
//   Phase 3: LayerNorm rows in-place in d_out
// Two n-tiles per iteration (dual accumulator chains) + A-frag prefetch so
// WMMA issue overlaps the next fragment loads.
// ---------------------------------------------------------------------------
__global__ __launch_bounds__(256) void k_mlp(const float*  __restrict__ x,
                                             const __bf16* __restrict__ hbf,
                                             const __bf16* __restrict__ W1T,
                                             const __bf16* __restrict__ W2T,
                                             const float*  __restrict__ gamma,
                                             const float*  __restrict__ beta,
                                             float*        __restrict__ out) {
    __shared__ __bf16 hid[16 * H4];               // 64 KB hidden tile (bf16)
    const int m0   = blockIdx.x * 16;
    const int lane = threadIdx.x & 31;
    const int wave = threadIdx.x >> 5;            // 8 waves
    const int lc   = lane & 15;                   // fragment row/col index
    const int lh   = lane >> 4;                   // K-half select

    // ---- Phase 1: GEMM1 + GELU -> LDS ----
    const __bf16* brow = hbf + (size_t)(m0 + lc) * H;   // node row (B columns)
    for (int p = 0; p < 8; ++p) {                       // 8 pairs of n-tiles
        const int n0 = wave * 256 + p * 32;
        const __bf16* arow0 = W1T + (size_t)(n0 +      lc) * H;
        const __bf16* arow1 = W1T + (size_t)(n0 + 16 + lc) * H;
        v8f acc0 = {0.f, 0.f, 0.f, 0.f, 0.f, 0.f, 0.f, 0.f};
        v8f acc1 = {0.f, 0.f, 0.f, 0.f, 0.f, 0.f, 0.f, 0.f};
        v16bf a0 = load_a_frag(arow0, lh);
        v16bf a1 = load_a_frag(arow1, lh);
#pragma unroll
        for (int k0 = 0; k0 < H; k0 += 32) {
            v16bf na0 = a0, na1 = a1;
            if (k0 + 32 < H) {                          // prefetch next K step
                na0 = load_a_frag(arow0 + k0 + 32, lh);
                na1 = load_a_frag(arow1 + k0 + 32, lh);
            }
            v16bf b = *(const v16bf*)(brow + k0 + 16 * lh);
            acc0 = __builtin_amdgcn_wmma_f32_16x16x32_bf16(
                false, a0, false, b, (short)0, acc0, false, false);
            acc1 = __builtin_amdgcn_wmma_f32_16x16x32_bf16(
                false, a1, false, b, (short)0, acc1, false, false);
            a0 = na0; a1 = na1;
        }
        // D: lane -> node m = lc, values n = n0 + 8*lh + r. GELU(exact) + pack.
        v8bf g0, g1;
#pragma unroll
        for (int r = 0; r < 8; ++r) {
            float v0 = acc0[r];
            float v1 = acc1[r];
            g0[r] = (__bf16)(0.5f * v0 * (1.0f + erff(v0 * 0.70710678118654752f)));
            g1[r] = (__bf16)(0.5f * v1 * (1.0f + erff(v1 * 0.70710678118654752f)));
        }
        *(v8bf*)(&hid[lc * H4 + n0 +      8 * lh]) = g0;   // 16B ds_store
        *(v8bf*)(&hid[lc * H4 + n0 + 16 + 8 * lh]) = g1;
    }
    __syncthreads();

    // ---- Phase 2: GEMM2 + residual -> d_out (pre-LN) ----
    const __bf16* hrow = &hid[lc * H4];                 // node row in LDS
    for (int p = 0; p < 2; ++p) {                       // 2 pairs of n-tiles
        const int n0 = wave * 64 + p * 32;
        const __bf16* arow0 = W2T + (size_t)(n0 +      lc) * H4;
        const __bf16* arow1 = W2T + (size_t)(n0 + 16 + lc) * H4;
        v8f acc0 = {0.f, 0.f, 0.f, 0.f, 0.f, 0.f, 0.f, 0.f};
        v8f acc1 = {0.f, 0.f, 0.f, 0.f, 0.f, 0.f, 0.f, 0.f};
        v16bf a0 = load_a_frag(arow0, lh);
        v16bf a1 = load_a_frag(arow1, lh);
#pragma unroll 8
        for (int k0 = 0; k0 < H4; k0 += 32) {
            v16bf na0 = a0, na1 = a1;
            if (k0 + 32 < H4) {                         // prefetch next K step
                na0 = load_a_frag(arow0 + k0 + 32, lh);
                na1 = load_a_frag(arow1 + k0 + 32, lh);
            }
            v16bf b = *(const v16bf*)(hrow + k0 + 16 * lh);  // 32B ds_load
            acc0 = __builtin_amdgcn_wmma_f32_16x16x32_bf16(
                false, a0, false, b, (short)0, acc0, false, false);
            acc1 = __builtin_amdgcn_wmma_f32_16x16x32_bf16(
                false, a1, false, b, (short)0, acc1, false, false);
            a0 = na0; a1 = na1;
        }
#pragma unroll
        for (int q = 0; q < 2; ++q) {
            const v8f& acc = q ? acc1 : acc0;
            const size_t ro = (size_t)(m0 + lc) * H + n0 + 16 * q + 8 * lh;
            float4 x0 = *(const float4*)(x + ro);
            float4 x1 = *(const float4*)(x + ro + 4);
            float4 o0 = {acc[0] + x0.x, acc[1] + x0.y, acc[2] + x0.z, acc[3] + x0.w};
            float4 o1 = {acc[4] + x1.x, acc[5] + x1.y, acc[6] + x1.z, acc[7] + x1.w};
            *(float4*)(out + ro)     = o0;
            *(float4*)(out + ro + 4) = o1;
        }
    }
    __syncthreads();

    // ---- Phase 3: LayerNorm (2 rows per wave, wave32 shuffle reduction) ----
#pragma unroll
    for (int rr = 0; rr < 2; ++rr) {
        const int m = wave * 2 + rr;
        float* row = out + (size_t)(m0 + m) * H;
        float s = 0.f, s2 = 0.f;
#pragma unroll
        for (int c = lane; c < H; c += 32) {
            float v = row[c];
            s  += v;
            s2 += v * v;
        }
#pragma unroll
        for (int off = 16; off >= 1; off >>= 1) {
            s  += __shfl_xor(s,  off, 32);
            s2 += __shfl_xor(s2, off, 32);
        }
        const float mean = s * (1.0f / H);
        float var = s2 * (1.0f / H) - mean * mean;
        var = fmaxf(var, 0.f);
        const float rstd = rsqrtf(var + LN_EPS);
#pragma unroll
        for (int c = lane; c < H; c += 32) {
            float v = row[c];
            row[c] = (v - mean) * rstd * gamma[c] + beta[c];
        }
    }
}

// ---------------------------------------------------------------------------
extern "C" void kernel_launch(void* const* d_in, const int* in_sizes, int n_in,
                              void* d_out, int out_size, void* d_ws, size_t ws_size,
                              hipStream_t stream) {
    const float* x     = (const float*)d_in[0];
    const float* ea    = (const float*)d_in[1];
    const float* W1    = (const float*)d_in[2];
    const float* W2    = (const float*)d_in[3];
    const float* gamma = (const float*)d_in[4];
    const float* beta  = (const float*)d_in[5];
    const int*   ei    = (const int*)d_in[6];
    float*       out   = (float*)d_out;

    // Workspace layout
    char* ws = (char*)d_ws;
    float*  aggr = (float*)ws;                                        // N*H f32
    __bf16* hbf  = (__bf16*)(ws + (size_t)N_NODES * H * 4);           // N*H bf16
    __bf16* W1T  = (__bf16*)(ws + (size_t)N_NODES * H * 4
                                + (size_t)N_NODES * H * 2);           // [2048][512]
    __bf16* W2T  = W1T + (size_t)H * H4;                              // [512][2048]

    hipMemsetAsync(aggr, 0, (size_t)N_NODES * H * sizeof(float), stream);

    {
        long total = (long)N_EDGES * (H / 4);
        k_scatter<<<(int)((total + 255) / 256), 256, 0, stream>>>(x, ea, ei, aggr);
    }
    {
        long total = (long)N_NODES * (H / 4);
        k_makeh<<<(int)((total + 255) / 256), 256, 0, stream>>>(x, aggr, hbf);
    }
    {
        long total = (long)H * H4;
        k_packT<<<(int)((total + 255) / 256), 256, 0, stream>>>(W1, W1T, H, H4);
        k_packT<<<(int)((total + 255) / 256), 256, 0, stream>>>(W2, W2T, H4, H);
    }
    k_mlp<<<N_NODES / 16, 256, 0, stream>>>(x, hbf, W1T, W2T, gamma, beta, out);
}
